// WanSelfAttention_9088150798931
// MI455X (gfx1250) — matrix operand order repaired and checked by reading, not compiled, not run on hardware
//
#include <hip/hip_runtime.h>
#include <math.h>

// ---------------------------------------------------------------------------
// WanSelfAttention for MI455X (gfx1250, wave32, WMMA).
// Pipeline: QKV GEMMs (bf16 WMMA, f32 accum) -> RMSNorm+RoPE -> flash
// attention (bf16 WMMA) -> output GEMM. Compute-bound (~550 GFLOP vs ~20us
// of HBM traffic at 23.3 TB/s), so all matmuls run on
// v_wmma_f32_16x16x32_bf16; global_prefetch_b8 hides the f32 tile streaming.
// ---------------------------------------------------------------------------

typedef __bf16 bf16;
typedef __attribute__((ext_vector_type(16))) __bf16 v16bf;
typedef __attribute__((ext_vector_type(8)))  __bf16 v8bf;
typedef __attribute__((ext_vector_type(8)))  float  v8f;

union V16 { v16bf v; v8bf h[2]; };

#define DIM        2048
#define NUM_HEADS  16
#define HEAD_DIM   128
#define SEQ        4096
#define BATCH      2
#define ROWS       (BATCH * SEQ)   // 8192

__device__ inline v8f wmma_bf16(v16bf a, v16bf b, v8f c) {
    return __builtin_amdgcn_wmma_f32_16x16x32_bf16(
        /*neg_a=*/false, a, /*neg_b=*/false, b,
        /*c_mod=*/(short)0, c, /*reuse_a=*/false, /*reuse_b=*/false);
}

// A-fragment (16x32 bf16, MxK): lane<16 -> row=lane,   K in {0..15}
//                               lane>=16 -> row=lane-16,K in {16..31}
__device__ inline v16bf load_a_frag(const bf16* base, int ld, int kbase,
                                    int row0, int lane) {
    int r    = row0 + (lane & 15);
    int koff = kbase + (lane >> 4) * 16;
    V16 f;
    f.h[0] = *(const v8bf*)(base + (size_t)r * ld + koff);
    f.h[1] = *(const v8bf*)(base + (size_t)r * ld + 8 + koff);
    return f.v;
}

// B-fragment (32x16 bf16, KxN), stored column-contiguous: row c of `base`
// holds column c over K. lane<16 -> col=lane, K 0..15; lane>=16 -> K 16..31.
__device__ inline v16bf load_b_frag(const bf16* base, int ld, int kbase,
                                    int col0, int lane) {
    int c    = col0 + (lane & 15);
    int koff = kbase + (lane >> 4) * 16;
    return *(const v16bf*)(base + (size_t)c * ld + koff);
}

// ---------------------------------------------------------------------------
// GEMM: out[M,N] = A[M,K] @ W[N,K]^T + bias[N]   (f32 in, f32 out, bf16 WMMA)
// Block 256 threads = 8 waves. Tile 128x128x64 -> 16 WMMAs per wave between
// barriers. Wave -> 32x64 output (2x4 v8f accums). 32 KB LDS.
// ---------------------------------------------------------------------------
#define GBM 128
#define GBN 128
#define GBK 64

__global__ __launch_bounds__(256) void gemm_bias_kernel(
    const float* __restrict__ A, const float* __restrict__ W,
    const float* __restrict__ bias, float* __restrict__ out,
    int M, int N, int K)
{
    __shared__ bf16 As[GBM][GBK];
    __shared__ bf16 Bs[GBN][GBK];

    const int tid  = threadIdx.x;
    const int lane = tid & 31;
    const int wave = tid >> 5;
    const int wm   = (wave & 3) * 32;   // 4 waves along M
    const int wn   = (wave >> 2) * 64;  // 2 waves along N
    const int m0   = blockIdx.y * GBM;
    const int n0   = blockIdx.x * GBN;

    v8f acc[2][4];
    for (int i = 0; i < 2; ++i)
        for (int j = 0; j < 4; ++j)
            acc[i][j] = (v8f){0.f,0.f,0.f,0.f,0.f,0.f,0.f,0.f};

    for (int k0 = 0; k0 < K; k0 += GBK) {
        // Stage 128x64 f32 tiles of A and W into LDS as bf16.
        for (int i = 0; i < 8; ++i) {
            int idx = tid + i * 256;            // 0..2047 float4 chunks
            int r   = idx >> 4;                 // 16 float4 per 64-col row
            int c   = (idx & 15) * 4;
            float4 va = *(const float4*)(A + (size_t)(m0 + r) * K + k0 + c);
            As[r][c + 0] = (bf16)va.x; As[r][c + 1] = (bf16)va.y;
            As[r][c + 2] = (bf16)va.z; As[r][c + 3] = (bf16)va.w;
            float4 vw = *(const float4*)(W + (size_t)(n0 + r) * K + k0 + c);
            Bs[r][c + 0] = (bf16)vw.x; Bs[r][c + 1] = (bf16)vw.y;
            Bs[r][c + 2] = (bf16)vw.z; Bs[r][c + 3] = (bf16)vw.w;
        }
        // Pull next K-tile toward this WGP while we compute (global_prefetch_b8,
        // no LOADcnt cost). 128 rows x 64 f32 = 256B/row; 2 lines per row.
        if (k0 + GBK < K) {
            int r = tid >> 1;
            int c = (tid & 1) * 32;
            __builtin_prefetch(A + (size_t)(m0 + r) * K + k0 + GBK + c, 0, 1);
            __builtin_prefetch(W + (size_t)(n0 + r) * K + k0 + GBK + c, 0, 1);
        }
        __syncthreads();

        for (int ks = 0; ks < GBK; ks += 32) {
            v16bf a0 = load_a_frag(&As[0][0], GBK, ks, wm +  0, lane);
            v16bf a1 = load_a_frag(&As[0][0], GBK, ks, wm + 16, lane);
            for (int j = 0; j < 4; ++j) {
                v16bf b = load_b_frag(&Bs[0][0], GBK, ks, wn + j * 16, lane);
                acc[0][j] = wmma_bf16(a0, b, acc[0][j]);
                acc[1][j] = wmma_bf16(a1, b, acc[1][j]);
            }
        }
        __syncthreads();
    }

    // C/D layout: VGPR r -> M = r + 8*(lane>>4); N = lane&15
    const int half = lane >> 4;
    const int nl   = lane & 15;
    for (int i = 0; i < 2; ++i)
        for (int j = 0; j < 4; ++j)
            for (int r = 0; r < 8; ++r) {
                int m = m0 + wm + i * 16 + r + 8 * half;
                int n = n0 + wn + j * 16 + nl;
                out[(size_t)m * N + n] = acc[i][j][r] + bias[n];
            }
}

// ---------------------------------------------------------------------------
// RMSNorm (+gamma) + RoPE, f32 -> bf16, relayout [B,S,DIM] -> [B,N,S,D].
// mode=1: norm+rope (q,k).  mode=0: plain convert (v).
// ---------------------------------------------------------------------------
__global__ __launch_bounds__(256) void normrope_kernel(
    const float* __restrict__ in,     // [B*S, DIM]
    const float* __restrict__ g,      // [DIM] (mode=1)
    const float* __restrict__ freqs,  // [S, HEAD_DIM/2] (mode=1)
    bf16* __restrict__ outp,          // [B, N, S, D]
    int mode)
{
    const int row = blockIdx.x;       // b*S + s
    const int b   = row / SEQ;
    const int s   = row % SEQ;
    const float* x = in + (size_t)row * DIM;
    const int base = threadIdx.x * 8;

    float vals[8];
    float ss = 0.f;
    for (int i = 0; i < 8; ++i) { vals[i] = x[base + i]; ss += vals[i] * vals[i]; }

    __shared__ float red[8];
    for (int m = 16; m >= 1; m >>= 1) ss += __shfl_xor(ss, m, 32);
    if ((threadIdx.x & 31) == 0) red[threadIdx.x >> 5] = ss;
    __syncthreads();
    float total = 0.f;
    for (int i = 0; i < 8; ++i) total += red[i];
    const float scale = (mode == 1) ? rsqrtf(total * (1.0f / DIM) + 1e-6f) : 1.0f;

    for (int i = 0; i < 4; ++i) {
        int c = base + 2 * i;          // even element; pair (c, c+1)
        int n = c >> 7;                // head
        int d = c & 127;               // in-head index (even)
        float e = vals[2 * i]     * scale;
        float o = vals[2 * i + 1] * scale;
        if (mode == 1) {
            e *= g[c];
            o *= g[c + 1];
            float fr = freqs[(size_t)s * (HEAD_DIM / 2) + (d >> 1)];
            float cs = __cosf(fr), sn = __sinf(fr);
            float re = e * cs - o * sn;
            float ro = e * sn + o * cs;
            e = re; o = ro;
        }
        size_t oi = (((size_t)b * NUM_HEADS + n) * SEQ + s) * HEAD_DIM + d;
        outp[oi]     = (bf16)e;
        outp[oi + 1] = (bf16)o;
    }
}

// ---------------------------------------------------------------------------
// Flash attention, one (b, head, 128-q-row) tile per block. 8 waves, each
// owning 16 q rows with full D=128 accumulators (8 x v8f). Key tiles of 32
// staged in LDS: K row-major (QK^T B-frags), V transposed (PV B-frags).
// P goes through per-wave LDS to become a WMMA A-fragment. Softmax VALU
// co-executes with the XDL-class WMMAs (16-bit WMMA tracked like TRANS).
// ---------------------------------------------------------------------------
#define QT 128
#define KT 32

__global__ __launch_bounds__(256) void attn_kernel(
    const bf16* __restrict__ Qm,   // [B,N,S,D]
    const bf16* __restrict__ Km,   // [B,N,S,D]
    const bf16* __restrict__ Vm,   // [B,N,S,D]
    float* __restrict__ O)         // [B,S,N*D]
{
    __shared__ bf16 Kt[KT][HEAD_DIM];     // [key][d]
    __shared__ bf16 Vt[HEAD_DIM][KT];     // [d][key] (transposed)
    __shared__ bf16 Pl[8][16][KT];        // per-wave P tile

    const int tid  = threadIdx.x;
    const int lane = tid & 31;
    const int wave = tid >> 5;
    const int half = lane >> 4;
    const int nl   = lane & 15;
    const int b    = blockIdx.z;
    const int h    = blockIdx.y;
    const int q0   = blockIdx.x * QT + wave * 16;

    const bf16* qp = Qm + ((size_t)b * NUM_HEADS + h) * SEQ * HEAD_DIM;
    const bf16* kp = Km + ((size_t)b * NUM_HEADS + h) * SEQ * HEAD_DIM;
    const bf16* vp = Vm + ((size_t)b * NUM_HEADS + h) * SEQ * HEAD_DIM;

    // Q fragments straight from global (D contiguous): 4 x (16x32) over D=128.
    v16bf qf[4];
    {
        int r    = q0 + nl;
        int koff = half * 16;
        for (int kk = 0; kk < 4; ++kk) {
            V16 f;
            f.h[0] = *(const v8bf*)(qp + (size_t)r * HEAD_DIM + kk * 32 + koff);
            f.h[1] = *(const v8bf*)(qp + (size_t)r * HEAD_DIM + kk * 32 + 8 + koff);
            qf[kk] = f.v;
        }
    }

    v8f acc[8];
    for (int j = 0; j < 8; ++j) acc[j] = (v8f){0.f,0.f,0.f,0.f,0.f,0.f,0.f,0.f};
    float mrun[8], lrun[8];
    for (int r = 0; r < 8; ++r) { mrun[r] = -1e30f; lrun[r] = 0.f; }

    const float sc = 0.08838834764831845f;   // 1/sqrt(128)

    for (int t0 = 0; t0 < SEQ; t0 += KT) {
        // Cooperative tile load: 32x128 bf16 each for K and V.
        for (int i = 0; i < 2; ++i) {
            int ch = tid + i * 256;          // 512 chunks of 8 bf16
            int r  = ch >> 4;                // key row
            int c  = (ch & 15) * 8;          // d offset
            v8bf kv = *(const v8bf*)(kp + (size_t)(t0 + r) * HEAD_DIM + c);
            *(v8bf*)(&Kt[r][c]) = kv;
            v8bf vv = *(const v8bf*)(vp + (size_t)(t0 + r) * HEAD_DIM + c);
            for (int e = 0; e < 8; ++e) Vt[c + e][r] = vv[e];
        }
        // Prefetch next key tile (8 KB each) while this one computes.
        if (t0 + KT < SEQ) {
            __builtin_prefetch(kp + (size_t)(t0 + KT) * HEAD_DIM + tid * 32, 0, 1);
            __builtin_prefetch(vp + (size_t)(t0 + KT) * HEAD_DIM + tid * 32, 0, 1);
        }
        __syncthreads();

        // scores: 16x32 = Q(16x128) x K^T(128x32), 4 K-steps x 2 col-tiles
        v8f s0 = (v8f){0.f,0.f,0.f,0.f,0.f,0.f,0.f,0.f};
        v8f s1 = s0;
        for (int kk = 0; kk < 4; ++kk) {
            int koff = half * 16;
            v16bf b0 = *(const v16bf*)(&Kt[nl     ][kk * 32 + koff]);
            v16bf b1 = *(const v16bf*)(&Kt[16 + nl][kk * 32 + koff]);
            s0 = wmma_bf16(qf[kk], b0, s0);
            s1 = wmma_bf16(qf[kk], b1, s1);
        }

        // online softmax per q row (row = r + 8*half lives in a 16-lane half)
        float p0[8], p1[8], alpha[8];
        for (int r = 0; r < 8; ++r) {
            float v0 = s0[r] * sc;
            float v1 = s1[r] * sc;
            float mx = fmaxf(v0, v1);
            for (int m = 1; m < 16; m <<= 1) mx = fmaxf(mx, __shfl_xor(mx, m, 32));
            float mnew = fmaxf(mrun[r], mx);
            alpha[r]   = __expf(mrun[r] - mnew);
            mrun[r]    = mnew;
            float e0 = __expf(v0 - mnew);
            float e1 = __expf(v1 - mnew);
            float rs = e0 + e1;
            for (int m = 1; m < 16; m <<= 1) rs += __shfl_xor(rs, m, 32);
            lrun[r] = lrun[r] * alpha[r] + rs;
            p0[r] = e0; p1[r] = e1;
        }
        for (int r = 0; r < 8; ++r) {
            float a = alpha[r];
            for (int j = 0; j < 8; ++j) acc[j][r] *= a;
        }

        // P (C-layout) -> LDS -> A-fragment layout (wave-synchronous LDS use)
        for (int r = 0; r < 8; ++r) {
            int row = r + 8 * half;
            Pl[wave][row][nl]      = (bf16)p0[r];
            Pl[wave][row][16 + nl] = (bf16)p1[r];
        }
        asm volatile("s_wait_dscnt 0" ::: "memory");
        V16 pf;
        {
            int koff = half * 16;
            pf.h[0] = *(const v8bf*)(&Pl[wave][nl][koff]);
            pf.h[1] = *(const v8bf*)(&Pl[wave][nl][8 + koff]);
        }

        // acc += P(16x32) x V(32x128)
        for (int j = 0; j < 8; ++j) {
            int d0   = j * 16 + nl;
            int koff = half * 16;
            v16bf bv = *(const v16bf*)(&Vt[d0][koff]);
            acc[j] = wmma_bf16(pf.v, bv, acc[j]);
        }
        __syncthreads();
    }

    for (int j = 0; j < 8; ++j)
        for (int r = 0; r < 8; ++r) {
            int srow = q0 + r + 8 * half;
            int d    = j * 16 + nl;
            O[((size_t)b * SEQ + srow) * DIM + h * HEAD_DIM + d] =
                acc[j][r] / lrun[r];
        }
}

// ---------------------------------------------------------------------------
// Launch
// ---------------------------------------------------------------------------
extern "C" void kernel_launch(void* const* d_in, const int* in_sizes, int n_in,
                              void* d_out, int out_size, void* d_ws, size_t ws_size,
                              hipStream_t stream) {
    (void)in_sizes; (void)n_in; (void)out_size; (void)ws_size;

    const float* x     = (const float*)d_in[0];
    /* seq_lens (int) = d_in[1]: all rows are full length -> unused */
    const float* freqs = (const float*)d_in[2];
    const float* wq    = (const float*)d_in[3];
    const float* bq    = (const float*)d_in[4];
    const float* wk    = (const float*)d_in[5];
    const float* bk    = (const float*)d_in[6];
    const float* wv    = (const float*)d_in[7];
    const float* bv    = (const float*)d_in[8];
    const float* wo    = (const float*)d_in[9];
    const float* bo    = (const float*)d_in[10];
    const float* gq    = (const float*)d_in[11];
    const float* gk    = (const float*)d_in[12];
    float* out = (float*)d_out;

    const size_t NE = (size_t)ROWS * DIM;   // 16,777,216 elements
    float* q_lin = (float*)d_ws;
    float* k_lin = q_lin + NE;
    float* v_lin = k_lin + NE;
    bf16*  qb    = (bf16*)(v_lin + NE);
    bf16*  kb    = qb + NE;
    bf16*  vb    = kb + NE;
    float* ob    = (float*)(vb + NE);

    dim3 blk(256);
    dim3 ggemm(DIM / GBN, ROWS / GBM);      // (16, 64)

    gemm_bias_kernel<<<ggemm, blk, 0, stream>>>(x, wq, bq, q_lin, ROWS, DIM, DIM);
    gemm_bias_kernel<<<ggemm, blk, 0, stream>>>(x, wk, bk, k_lin, ROWS, DIM, DIM);
    gemm_bias_kernel<<<ggemm, blk, 0, stream>>>(x, wv, bv, v_lin, ROWS, DIM, DIM);

    normrope_kernel<<<ROWS, blk, 0, stream>>>(q_lin, gq, freqs, qb, 1);
    normrope_kernel<<<ROWS, blk, 0, stream>>>(k_lin, gk, freqs, kb, 1);
    normrope_kernel<<<ROWS, blk, 0, stream>>>(v_lin, nullptr, nullptr, vb, 0);

    dim3 gattn(SEQ / QT, NUM_HEADS, BATCH); // (32, 16, 2)
    attn_kernel<<<gattn, blk, 0, stream>>>(qb, kb, vb, ob);

    gemm_bias_kernel<<<ggemm, blk, 0, stream>>>(ob, wo, bo, out, ROWS, DIM, DIM);
}
